// DynamicFeatureAggregation_76184129896956
// MI455X (gfx1250) — matrix operand order repaired and checked by reading, not compiled, not run on hardware
//
#include <hip/hip_runtime.h>
#include <hip/hip_bf16.h>

typedef __attribute__((ext_vector_type(16))) _Float16 v16h;
typedef __attribute__((ext_vector_type(8)))  _Float16 v8h;
typedef __attribute__((ext_vector_type(2)))  _Float16 h2;
typedef __attribute__((ext_vector_type(8)))  float    v8f;

#define CCH 256
#define HH  64
#define WWD 64
#define HW  4096
#define BNE 1e-5f
#define AS  40   // LDS A-tile row stride in halves: 80B = 16B-aligned, conflict-free

// ---------------- prep: w -> f16, fold conv-bias + eval-BN into scale/bias ----
__global__ void dfa_prep(const float* __restrict__ w, const float* __restrict__ bsrc,
                         const float* __restrict__ gamma, const float* __restrict__ beta,
                         const float* __restrict__ mean, const float* __restrict__ var,
                         _Float16* __restrict__ w16, float* __restrict__ scale,
                         float* __restrict__ bias) {
  int i = blockIdx.x * 256 + threadIdx.x;
  if (i < CCH * 2 * CCH) w16[i] = (_Float16)w[i];
  if (i < CCH) {
    float inv = gamma[i] * rsqrtf(var[i] + BNE);
    scale[i] = inv;
    bias[i]  = bsrc[i] * inv + beta[i] - mean[i] * inv;
  }
}

// Stage one 32-channel K-chunk (raw x for kc<8, 3x3 sim-aggregation for kc>=8)
// into an LDS A-tile [64 px][AS halves], f16, pixel-major.
__device__ __forceinline__ void stage_chunk(int kc, _Float16* __restrict__ abuf,
                                            const float* __restrict__ xb,
                                            const float* __restrict__ sim_s,
                                            int h0, int hw0, int cp, int li) {
  if (kc < 8) {
    const int c0 = 32 * kc + 2 * cp;
    const float* p0 = xb + (size_t)c0 * HW + hw0;
    const float* p1 = p0 + HW;
    if (kc < 7) __builtin_prefetch(p0 + 32 * HW, 0, 0);   // global_prefetch next chunk
#pragma unroll
    for (int i = 0; i < 4; ++i) {
      int px = li + 16 * i;                  // lanes -> consecutive addrs (coalesced)
      h2 v; v.x = (_Float16)p0[px]; v.y = (_Float16)p1[px];
      *(h2*)(abuf + px * AS + 2 * cp) = v;   // bank-conflict-free (20*li is a perm)
    }
  } else {
    const int c0 = 32 * (kc - 8) + 2 * cp;
    const float* p0 = xb + (size_t)c0 * HW;
    const float* p1 = p0 + HW;
#pragma unroll
    for (int i = 0; i < 4; ++i) {
      int px = li + 16 * i;                  // px == w coordinate (single-row tile)
      float a0 = 0.f, a1 = 0.f;
#pragma unroll
      for (int dy = 0; dy < 3; ++dy) {
        int y = h0 + dy - 1;
        if ((unsigned)y < HH) {
#pragma unroll
          for (int dx = 0; dx < 3; ++dx) {
            int xw = px + dx - 1;
            if ((unsigned)xw < WWD) {
              float sv = sim_s[(dy * 3 + dx) * 64 + px];
              a0 += sv * p0[y * WWD + xw];
              a1 += sv * p1[y * WWD + xw];
            }
          }
        }
      }
      h2 v; v.x = (_Float16)a0; v.y = (_Float16)a1;
      *(h2*)(abuf + px * AS + 2 * cp) = v;
    }
  }
}

// ---------------- fused: aggregate(3x3,sim) + concat + 1x1-conv GEMM + BN + ReLU
__global__ __launch_bounds__(256)
void dfa_main(const float* __restrict__ x, const float* __restrict__ sim,
              const _Float16* __restrict__ w16, const float* __restrict__ scale_g,
              const float* __restrict__ bias_g, float* __restrict__ out) {
  __shared__ __align__(16) unsigned char smem[17408];
  float*    sim_s = (float*)smem;              // [9][64] during GEMM phase
  _Float16* a_s0  = (_Float16*)(smem + 2304);  // [64][AS] f16 A-tile, buffer 0
  _Float16* a_s1  = (_Float16*)(smem + 7424);  // [64][AS] f16 A-tile, buffer 1
  float*    c_s   = (float*)smem;              // [64][68] epilogue transpose (reuse)

  const int tid = threadIdx.x;
  const int g   = blockIdx.x;
  const int b   = g >> 6;        // batch
  const int h0  = g & 63;        // image row handled by this workgroup
  const int hw0 = h0 * WWD;

  // load similarity tile for this row: 9 x 64
  for (int f = tid; f < 9 * 64; f += 256) {
    int j = f >> 6, px = f & 63;
    sim_s[f] = sim[(size_t)b * 9 * HW + (size_t)j * HW + hw0 + px];
  }

  const int lane = tid & 31;
  const int wv   = tid >> 5;     // 8 waves: 4 M-subtiles x 2 N-halves
  const int l    = lane & 15;
  const int hi   = lane >> 4;
  const int m0   = (wv >> 1) * 16;
  const int nh   = wv & 1;

  // staging thread mapping: channel pair (cp), pixel lane (li)
  const int cp = tid >> 4;
  const int li = tid & 15;

  v8f acc[8];
#pragma unroll
  for (int t = 0; t < 8; ++t) acc[t] = (v8f){0,0,0,0,0,0,0,0};

  const float* xb = x + (size_t)b * CCH * HW;

  // prologue: stage chunk 0 into buffer 0 (no sim needed for x-chunks)
  stage_chunk(0, a_s0, xb, sim_s, h0, hw0, cp, li);
  __syncthreads();

  for (int kc = 0; kc < 16; ++kc) {
    _Float16* cur = (kc & 1) ? a_s1 : a_s0;
    _Float16* nxt = (kc & 1) ? a_s0 : a_s1;

    // --- issue all 8 B fragments first (one clause of 16 global_load_b128) ---
    const _Float16* bbase = w16 + (size_t)(128 * nh + l) * 512 + 32 * kc + hi * 16;
    v16h bfrag[8];
#pragma unroll
    for (int t = 0; t < 8; ++t)
      bfrag[t] = *(const v16h*)(bbase + (size_t)t * 16 * 512);

    // --- A fragment from current LDS buffer ---
    const _Float16* ap = cur + (m0 + l) * AS + hi * 16;
    v8h alo = *(const v8h*)ap;          // ds_load_b128, 16B-aligned
    v8h ah8 = *(const v8h*)(ap + 8);
    v16h afrag;
#pragma unroll
    for (int q = 0; q < 8; ++q) { afrag[q] = alo[q]; afrag[q + 8] = ah8[q]; }

    // --- stage next chunk into the other buffer; overlaps with WMMA below ---
    if (kc < 15) stage_chunk(kc + 1, nxt, xb, sim_s, h0, hw0, cp, li);

    // --- 8 WMMAs, waits on bfrag loads are staggered, staging is in flight ---
#pragma unroll
    for (int t = 0; t < 8; ++t)
      acc[t] = __builtin_amdgcn_wmma_f32_16x16x32_f16(false, afrag, false, bfrag[t],
                                                      (short)0, acc[t], false, false);

    __syncthreads();   // staging of kc+1 done; cur safe to rewrite at kc+1
  }

  // ---- epilogue: BN-fused scale/bias + ReLU, LDS transpose, coalesced stores
  for (int cc = 0; cc < 4; ++cc) {        // 4 chunks of 64 output channels
    __syncthreads();
    if (nh == (cc >> 1)) {
      int jbase = (cc & 1) * 4;
#pragma unroll
      for (int j2 = 0; j2 < 4; ++j2) {
        int t  = jbase + j2;              // acc tile; global tile = 8*nh + t
        int og = 64 * cc + 16 * j2 + l;   // global output channel
        float sc = scale_g[og];
        float bs = bias_g[og];
        v8f a = acc[t];
#pragma unroll
        for (int i = 0; i < 8; ++i) {
          int px = m0 + i + 8 * hi;       // C layout: VGPR i -> M = i (+8 upper lanes)
          float v2 = a[i] * sc + bs;
          c_s[(16 * j2 + l) * 68 + px] = fmaxf(v2, 0.f);
        }
      }
    }
    __syncthreads();
    float* obase = out + ((size_t)b * CCH + 64 * cc) * HW + hw0;
#pragma unroll
    for (int r = 0; r < 16; ++r) {        // 64ch x 64px, 256B coalesced runs
      int f = tid + 256 * r;
      int o = f >> 6, px = f & 63;
      obase[(size_t)o * HW + px] = c_s[o * 68 + px];
    }
  }
}

extern "C" void kernel_launch(void* const* d_in, const int* in_sizes, int n_in,
                              void* d_out, int out_size, void* d_ws, size_t ws_size,
                              hipStream_t stream) {
  const float* x     = (const float*)d_in[0];
  const float* sim   = (const float*)d_in[1];
  const float* w     = (const float*)d_in[2];
  const float* bconv = (const float*)d_in[3];
  const float* gamma = (const float*)d_in[4];
  const float* beta  = (const float*)d_in[5];
  const float* rmean = (const float*)d_in[6];
  const float* rvar  = (const float*)d_in[7];
  float* out = (float*)d_out;

  _Float16* w16   = (_Float16*)d_ws;                       // 256*512 f16 = 256KB
  float*    scale = (float*)((char*)d_ws + 262144);        // 256 f32
  float*    bias  = scale + 256;                           // 256 f32

  dfa_prep<<<512, 256, 0, stream>>>(w, bconv, gamma, beta, rmean, rvar,
                                    w16, scale, bias);
  // 8 batches x 64 rows = 512 workgroups, 256 threads (8 wave32 waves)
  dfa_main<<<512, 256, 0, stream>>>(x, sim, w16, scale, bias, out);
}
